// SEG_NET_37477884625010
// MI455X (gfx1250) — matrix operand order repaired and checked by reading, not compiled, not run on hardware
//
#include <hip/hip_runtime.h>
#include <hip/hip_bf16.h>
#include <math.h>

typedef __attribute__((ext_vector_type(16))) _Float16 v16h;
typedef __attribute__((ext_vector_type(8)))  float    v8f;

#define NBOX 64

// Per-lane async global->LDS copy (CDNA5 GLOBAL_LOAD_ASYNC_TO_LDS_B32, ASYNCcnt).
// No VGPR round-trip: 4 bytes land directly at this lane's LDS address.
__device__ __forceinline__ void async_ld_f32(float* lds, const float* g)
{
  unsigned la = (unsigned)(size_t)(void*)lds;            // flat shared addr[31:0] == LDS byte offset
  unsigned long long ga = (unsigned long long)(size_t)(const void*)g;
  asm volatile("global_load_async_to_lds_b32 %0, %1, off" :: "v"(la), "v"(ga) : "memory");
}
__device__ __forceinline__ void wait_async0()
{
  asm volatile("s_wait_asynccnt 0" ::: "memory");
}

// ------------------------------------------------------------------
// Implicit-GEMM convolution on the WMMA pipe (KS = 1 or 3, square pow2 W).
// Each wave32 computes a 16(outch) x 32(pixel) tile: one A fragment feeds
// two v_wmma_f32_16x16x32_f16 per 32-wide K step. Staging tiles are
// per-wave LDS filled by async global->LDS DMA (border/K-tail zeros via a
// rare ds_store under the complementary exec mask); only wave-level
// ordering is needed -- no block barriers.
// ------------------------------------------------------------------
template<int KS>
__global__ __launch_bounds__(256)
void conv_wmma_kernel(const float* __restrict__ in1, const float* __restrict__ in2,
                      const float* __restrict__ wgt, const float* __restrict__ bias,
                      float* __restrict__ out,
                      int Cin1, int Cin, int Cout, int H, int logW, int relu)
{
  const int W    = 1 << logW;
  const int HW   = H << logW;
  const int lane = threadIdx.x;          // 0..31
  const int wv   = threadIdx.y;          // 0..7 (wave in block)
  const int half = lane >> 4;
  const int lmod = lane & 15;
  const int nTiles = HW >> 5;            // 32-pixel tiles (HW % 32 == 0 for all layers)
  int ptile = blockIdx.x * 8 + wv;
  const bool active = (ptile < nTiles);
  if (!active) ptile = nTiles - 1;
  const int b     = blockIdx.z;
  const int co0   = blockIdx.y << 4;
  const int Ktot  = Cin * KS * KS;
  const int Cin2  = Cin - Cin1;
  const int pbase = ptile << 5;

  __shared__ float lA[8][16 * 32];       // per-wave [m][k] weight tile (f32)
  __shared__ float lB[8][32 * 32];       // per-wave [k][n] im2col tile (f32)
  float* sA = lA[wv];
  float* sB = lB[wv];

  v8f acc0 = {}, acc1 = {};

  for (int k0 = 0; k0 < Ktot; k0 += 32) {
    // prefetch next weight K-slice into cache (global_prefetch_b8)
    if (k0 + 32 < Ktot)
      __builtin_prefetch(&wgt[(size_t)co0 * Ktot + k0 + 32 + lane], 0, 1);

    // ---- stage A: 16 outch x 32 k of weights (zero-pad K tail) ----
    #pragma unroll
    for (int t = lane; t < 512; t += 32) {
      int m  = t >> 5;
      int kk = t & 31;
      int k  = k0 + kk;
      if (k < Ktot) async_ld_f32(&sA[t], &wgt[(size_t)(co0 + m) * Ktot + k]);
      else          sA[t] = 0.f;
    }
    // ---- stage B: 32 k x 32 pixels, im2col gather with border zeros ----
    #pragma unroll
    for (int t = lane; t < 1024; t += 32) {
      int kk = t >> 5;
      int n  = t & 31;
      int k  = k0 + kk;
      int p  = pbase + n;
      int oy = p >> logW;
      int ox = p & (W - 1);
      int cin, iy, ix;
      if (KS == 1) {
        cin = k; iy = oy; ix = ox;
      } else {
        cin = k / (KS * KS);                    // constant divide -> magic mul
        int rs = k - cin * (KS * KS);
        int rr = rs / KS;
        iy = oy + rr - (KS >> 1);
        ix = ox + (rs - rr * KS) - (KS >> 1);
      }
      bool inb = (k < Ktot) &&
                 (KS == 1 || ((unsigned)iy < (unsigned)H && (unsigned)ix < (unsigned)W));
      if (inb) {
        const float* src = (cin < Cin1)
          ? &in1[((size_t)b * Cin1 + cin) * HW + ((size_t)iy << logW) + ix]
          : &in2[((size_t)b * Cin2 + (cin - Cin1)) * HW + ((size_t)iy << logW) + ix];
        async_ld_f32(&sB[t], src);
      } else {
        sB[t] = 0.f;
      }
    }
    wait_async0();                       // async DMA data visible in LDS
    __builtin_amdgcn_wave_barrier();     // order staging before DS reads (wave-local LDS)

    // ---- build fragments in contiguous k order, same for A and B (f32 -> f16) ----
    v16h afrag, bfrag0, bfrag1;
    #pragma unroll
    for (int j = 0; j < 16; ++j) {
      // A 16x32 f16: lane m=lane%16; element j -> k = half*16 + j
      int ka = (half << 4) + j;
      afrag[j] = (_Float16)sA[lmod * 32 + ka];
      // B 32x16 f16: lane n=lane%16; element j -> k = half*16 + j
      int kb = (half << 4) + j;
      bfrag0[j] = (_Float16)sB[kb * 32 + lmod];
      bfrag1[j] = (_Float16)sB[kb * 32 + 16 + lmod];
    }
    acc0 = __builtin_amdgcn_wmma_f32_16x16x32_f16(false, afrag, false, bfrag0,
                                                  (short)0, acc0, false, false);
    acc1 = __builtin_amdgcn_wmma_f32_16x16x32_f16(false, afrag, false, bfrag1,
                                                  (short)0, acc1, false, false);
    __builtin_amdgcn_wave_barrier();     // keep next staging after fragment reads
  }

  if (active) {
    int p0 = pbase + lmod;
    #pragma unroll
    for (int r = 0; r < 8; ++r) {
      int m  = r + (half << 3);          // C/D layout: vgpr r, lanes16-31 -> M+8
      int co = co0 + m;
      float bv = bias[co];
      float v0 = acc0[r] + bv;
      float v1 = acc1[r] + bv;
      if (relu) { v0 = fmaxf(v0, 0.f); v1 = fmaxf(v1, 0.f); }
      size_t base = ((size_t)b * Cout + co) * HW;
      out[base + p0]      = v0;
      out[base + p0 + 16] = v1;
    }
  }
}

// ------------------------------------------------------------------
// ROI bilinear crop (align_corners linspace over the box), NCHW.
// ------------------------------------------------------------------
__global__ void crop_kernel(const float* __restrict__ feat, const float* __restrict__ boxes,
                            const int* __restrict__ bb, float* __restrict__ out,
                            int C, int h, int w, int p)
{
  int idx = blockIdx.x * blockDim.x + threadIdx.x;
  int total = NBOX * C * p * p;
  if (idx >= total) return;
  int px = idx % p;
  int py = (idx / p) % p;
  int c  = (idx / (p * p)) % C;
  int n  = idx / (p * p * C);

  float y1 = boxes[n * 4 + 0], x1 = boxes[n * 4 + 1];
  float y2 = boxes[n * 4 + 2], x2 = boxes[n * 4 + 3];
  float ty = (p > 1) ? (float)py / (float)(p - 1) : 0.f;
  float tx = (p > 1) ? (float)px / (float)(p - 1) : 0.f;
  float ys = (y1 + (y2 - y1) * ty) * (float)(h - 1);
  float xs = (x1 + (x2 - x1) * tx) * (float)(w - 1);

  float y0f = floorf(ys); y0f = fminf(fmaxf(y0f, 0.f), (float)(h - 1));
  float x0f = floorf(xs); x0f = fminf(fmaxf(x0f, 0.f), (float)(w - 1));
  int y0 = (int)y0f, x0 = (int)x0f;
  int y1i = (y0 + 1 < h - 1) ? y0 + 1 : h - 1;
  int x1i = (x0 + 1 < w - 1) ? x0 + 1 : w - 1;
  float wy = ys - y0f, wx = xs - x0f;

  int bi = bb[n];
  const float* f = feat + ((size_t)bi * C + c) * h * w;
  float v00 = f[(size_t)y0  * w + x0 ];
  float v01 = f[(size_t)y0  * w + x1i];
  float v10 = f[(size_t)y1i * w + x0 ];
  float v11 = f[(size_t)y1i * w + x1i];
  out[idx] = v00 * (1.f - wy) * (1.f - wx) + v01 * (1.f - wy) * wx
           + v10 * wy * (1.f - wx)         + v11 * wy * wx;
}

// ------------------------------------------------------------------
// Bilinear upsample, align_corners=True. One thread per output elem.
// ------------------------------------------------------------------
__global__ void upsample_kernel(const float* __restrict__ in, float* __restrict__ out,
                                int NC, int h, int w, int oh, int ow)
{
  int idx = blockIdx.x * blockDim.x + threadIdx.x;
  int total = NC * oh * ow;
  if (idx >= total) return;
  int ox = idx % ow;
  int oy = (idx / ow) % oh;
  int nc = idx / (ow * oh);

  float ys = (oh > 1) ? (float)oy * (float)(h - 1) / (float)(oh - 1) : 0.f;
  float xs = (ow > 1) ? (float)ox * (float)(w - 1) / (float)(ow - 1) : 0.f;
  float y0f = floorf(ys), x0f = floorf(xs);
  int y0 = (int)y0f, x0 = (int)x0f;
  int y1 = (y0 + 1 < h - 1) ? y0 + 1 : h - 1;
  int x1 = (x0 + 1 < w - 1) ? x0 + 1 : w - 1;
  float wy = ys - y0f, wx = xs - x0f;

  const float* f = in + (size_t)nc * h * w;
  float v00 = f[(size_t)y0 * w + x0];
  float v01 = f[(size_t)y0 * w + x1];
  float v10 = f[(size_t)y1 * w + x0];
  float v11 = f[(size_t)y1 * w + x1];
  out[idx] = v00 * (1.f - wy) * (1.f - wx) + v01 * (1.f - wy) * wx
           + v10 * wy * (1.f - wx)         + v11 * wy * wx;
}

// ------------------------------------------------------------------
// CRM (SE attention): spatial max/mean pool, FC1, FC2+sigmoid, in-place scale
// ------------------------------------------------------------------
__global__ void pool_kernel(const float* __restrict__ x, float* __restrict__ mx,
                            float* __restrict__ av, int NC, int hw)
{
  int i = blockIdx.x * blockDim.x + threadIdx.x;
  if (i >= NC) return;
  const float* p = x + (size_t)i * hw;
  float m = p[0], s = 0.f;
  for (int e = 0; e < hw; ++e) { float v = p[e]; m = fmaxf(m, v); s += v; }
  mx[i] = m;
  av[i] = s / (float)hw;
}

__global__ void crm_hidden_kernel(const float* __restrict__ mx, const float* __restrict__ av,
                                  const float* __restrict__ w1, const float* __restrict__ b1,
                                  float* __restrict__ hmx, float* __restrict__ hav,
                                  int C, int Chid)
{
  int i = blockIdx.x * blockDim.x + threadIdx.x;
  if (i >= NBOX * Chid) return;
  int j = i % Chid, n = i / Chid;
  const float* wr = w1 + (size_t)j * C;
  const float* mr = mx + (size_t)n * C;
  const float* ar = av + (size_t)n * C;
  float sm = b1[j], sa = b1[j];
  for (int c = 0; c < C; ++c) { sm += mr[c] * wr[c]; sa += ar[c] * wr[c]; }
  hmx[i] = sm;
  hav[i] = sa;
}

__global__ void crm_scale_kernel(float* __restrict__ x,
                                 const float* __restrict__ hmx, const float* __restrict__ hav,
                                 const float* __restrict__ w2, const float* __restrict__ b2,
                                 int C, int Chid, int hw)
{
  int i = blockIdx.x * blockDim.x + threadIdx.x;
  if (i >= NBOX * C) return;
  int c = i % C, n = i / C;
  const float* wr = w2 + (size_t)c * Chid;
  const float* hm = hmx + (size_t)n * Chid;
  const float* ha = hav + (size_t)n * Chid;
  float s = 2.f * b2[c];
  for (int j = 0; j < Chid; ++j) s += (hm[j] + ha[j]) * wr[j];
  s = 1.f / (1.f + expf(-s));
  float* px = x + (size_t)i * hw;
  for (int e = 0; e < hw; ++e) px[e] *= s;
}

// ------------------------------------------------------------------
// Final 3x3 conv 64->1 + sigmoid into d_out [NBOX,64,64]
// ------------------------------------------------------------------
__global__ void final_conv_kernel(const float* __restrict__ x, const float* __restrict__ w,
                                  const float* __restrict__ b, float* __restrict__ out)
{
  int idx = blockIdx.x * blockDim.x + threadIdx.x;
  int total = NBOX * 64 * 64;
  if (idx >= total) return;
  int ox = idx & 63;
  int oy = (idx >> 6) & 63;
  int n  = idx >> 12;
  float acc = b[0];
  for (int c = 0; c < 64; ++c) {
    const float* xc = x + (((size_t)n * 64 + c) << 12);
    const float* wc = w + c * 9;
    #pragma unroll
    for (int r = -1; r <= 1; ++r) {
      #pragma unroll
      for (int s = -1; s <= 1; ++s) {
        int iy = oy + r, ix = ox + s;
        if (iy >= 0 && iy < 64 && ix >= 0 && ix < 64)
          acc += xc[(iy << 6) + ix] * wc[(r + 1) * 3 + (s + 1)];
      }
    }
  }
  out[idx] = 1.f / (1.f + expf(-acc));
}

// ------------------------------------------------------------------
extern "C" void kernel_launch(void* const* d_in, const int* in_sizes, int n_in,
                              void* d_out, int out_size, void* d_ws, size_t ws_size,
                              hipStream_t stream)
{
  (void)in_sizes; (void)n_in; (void)out_size; (void)ws_size;
  const float* boxes = (const float*)d_in[0];
  const int*   bb    = (const int*)  d_in[1];
  const float* feat0 = (const float*)d_in[2];
  const float* feat1 = (const float*)d_in[3];
  const float* feat2 = (const float*)d_in[4];
  const float* feat3 = (const float*)d_in[5];
  const float* feat4 = (const float*)d_in[6];
  int pi = 7;
  const float* c0_w1     = (const float*)d_in[pi++];
  const float* c0_b1     = (const float*)d_in[pi++];
  const float* c0_w2     = (const float*)d_in[pi++];
  const float* c0_b2     = (const float*)d_in[pi++];
  const float* sc0_up_w  = (const float*)d_in[pi++];
  const float* sc0_up_b  = (const float*)d_in[pi++];
  const float* sc0_cat_w = (const float*)d_in[pi++];
  const float* sc0_cat_b = (const float*)d_in[pi++];
  const float* sc1_up_w  = (const float*)d_in[pi++];
  const float* sc1_up_b  = (const float*)d_in[pi++];
  const float* sc1_cat_w = (const float*)d_in[pi++];
  const float* sc1_cat_b = (const float*)d_in[pi++];
  const float* sc2_up_w  = (const float*)d_in[pi++];
  const float* sc2_up_b  = (const float*)d_in[pi++];
  const float* sc2_cat_w = (const float*)d_in[pi++];
  const float* sc2_cat_b = (const float*)d_in[pi++];
  const float* sc3_up_w  = (const float*)d_in[pi++];
  const float* sc3_up_b  = (const float*)d_in[pi++];
  const float* sc3_cat_w = (const float*)d_in[pi++];
  const float* sc3_cat_b = (const float*)d_in[pi++];
  const float* fr_w1     = (const float*)d_in[pi++];
  const float* fr_b1     = (const float*)d_in[pi++];
  const float* fr_w2     = (const float*)d_in[pi++];
  const float* fr_b2     = (const float*)d_in[pi++];
  const float* fr_w3     = (const float*)d_in[pi++];
  const float* fr_b3     = (const float*)d_in[pi++];
  const float* crm512_w1  = (const float*)d_in[pi++];
  const float* crm512_b1  = (const float*)d_in[pi++];
  const float* crm512_w2  = (const float*)d_in[pi++];
  const float* crm512_b2  = (const float*)d_in[pi++];
  const float* crm1024_w1 = (const float*)d_in[pi++];
  const float* crm1024_b1 = (const float*)d_in[pi++];
  const float* crm1024_w2 = (const float*)d_in[pi++];
  const float* crm1024_b2 = (const float*)d_in[pi++];
  float* out = (float*)d_out;

  // ---- workspace layout (floats) ----
  float* ws = (float*)d_ws;
  const size_t SLAB = (size_t)64 * 64 * 64 * 64;   // 16,777,216 floats
  float* SA = ws;
  float* SB = ws + SLAB;
  float* SC = ws + 2 * SLAB;
  float* SD = ws + 3 * SLAB;
  float* SE = ws + 4 * SLAB;                       // 4,194,304 floats used
  float* mx  = SE + (size_t)64 * 64 * 32 * 32;
  float* av  = mx + 64 * 1024;
  float* hmx = av + 64 * 1024;
  float* hav = hmx + 64 * 128;

  auto conv = [&](const float* i1, const float* i2, const float* w, const float* bi,
                  float* o, int Cin1, int Cin, int Cout, int Wd, int ks,
                  int nb, int relu) {
    int logW = __builtin_ctz(Wd);                  // all spatial sizes are pow2, square
    int nTiles = (Wd * Wd) >> 5;
    dim3 grid((nTiles + 7) / 8, Cout >> 4, nb);
    dim3 block(32, 8, 1);
    if (ks == 3)
      conv_wmma_kernel<3><<<grid, block, 0, stream>>>(i1, i2 ? i2 : i1, w, bi, o,
                                                      Cin1, Cin, Cout, Wd, logW, relu);
    else
      conv_wmma_kernel<1><<<grid, block, 0, stream>>>(i1, i2 ? i2 : i1, w, bi, o,
                                                      Cin1, Cin, Cout, Wd, logW, relu);
  };
  auto crop = [&](const float* f, int C, int h, int w, int p, float* o) {
    int total = NBOX * C * p * p;
    crop_kernel<<<dim3((total + 255) / 256), dim3(256), 0, stream>>>(f, boxes, bb, o, C, h, w, p);
  };
  auto upsample = [&](const float* i, float* o, int NC, int h, int w, int oh, int ow) {
    int total = NC * oh * ow;
    upsample_kernel<<<dim3((total + 255) / 256), dim3(256), 0, stream>>>(i, o, NC, h, w, oh, ow);
  };
  auto crm = [&](float* x, int C, int Chid, int hw,
                 const float* w1, const float* b1, const float* w2, const float* b2) {
    int nc = NBOX * C;
    pool_kernel<<<dim3((nc + 255) / 256), dim3(256), 0, stream>>>(x, mx, av, nc, hw);
    int nh = NBOX * Chid;
    crm_hidden_kernel<<<dim3((nh + 255) / 256), dim3(256), 0, stream>>>(mx, av, w1, b1, hmx, hav, C, Chid);
    crm_scale_kernel<<<dim3((nc + 255) / 256), dim3(256), 0, stream>>>(x, hmx, hav, w2, b2, C, Chid, hw);
  };

  // ---- stem convs on full-res image (B=2) ----
  conv(feat0, nullptr, c0_w1, c0_b1, SA, 3, 3, 64, 256, 3, 2, 1);      // f0a
  conv(SA,    nullptr, c0_w2, c0_b2, SB, 64, 64, 64, 256, 3, 2, 1);    // f0b

  // ---- ROI crops ----
  crop(SB, 64, 256, 256, 64, SC);                           // p0 -> SC
  float* p1 = SA;                                           // SA reused
  float* p2 = p1 + (size_t)64 * 64 * 32 * 32;
  float* p3 = p2 + (size_t)64 * 256 * 16 * 16;
  float* p4 = p3 + (size_t)64 * 512 * 8 * 8;
  crop(feat1, 64,   128, 128, 32, p1);
  crop(feat2, 256,  64,  64,  16, p2);
  crop(feat3, 512,  32,  32,  8,  p3);
  crop(feat4, 1024, 16,  16,  4,  p4);

  // ---- CRM1024 on p4 (in place) ----
  crm(p4, 1024, 128, 16, crm1024_w1, crm1024_b1, crm1024_w2, crm1024_b2);

  // ---- comb level 3 ----
  upsample(p4, SB, 64 * 1024, 4, 4, 8, 8);
  conv(SB, nullptr, sc3_up_w, sc3_up_b, SD, 1024, 1024, 512, 8, 3, NBOX, 1);   // u3
  conv(p3, SD, sc3_cat_w, sc3_cat_b, SE, 512, 1024, 512, 8, 1, NBOX, 1);       // pre3
  crm(SE, 512, 64, 64, crm512_w1, crm512_b1, crm512_w2, crm512_b2);

  // ---- comb level 2 ----
  upsample(SE, SB, 64 * 512, 8, 8, 16, 16);
  conv(SB, nullptr, sc2_up_w, sc2_up_b, SD, 512, 512, 256, 16, 3, NBOX, 1);    // u2
  conv(p2, SD, sc2_cat_w, sc2_cat_b, SE, 256, 512, 256, 16, 1, NBOX, 1);       // pre2

  // ---- comb level 1 ----
  upsample(SE, SB, 64 * 256, 16, 16, 32, 32);
  conv(SB, nullptr, sc1_up_w, sc1_up_b, SD, 256, 256, 64, 32, 3, NBOX, 1);     // u1
  conv(p1, SD, sc1_cat_w, sc1_cat_b, SE, 64, 128, 64, 32, 1, NBOX, 1);         // pre1

  // ---- comb level 0 ----
  upsample(SE, SB, 64 * 64, 32, 32, 64, 64);
  conv(SB, nullptr, sc0_up_w, sc0_up_b, SD, 64, 64, 64, 64, 3, NBOX, 1);       // u0
  conv(SC, SD, sc0_cat_w, sc0_cat_b, SA, 64, 128, 64, 64, 1, NBOX, 1);         // pre0

  // ---- final refinement ----
  conv(SA, nullptr, fr_w1, fr_b1, SB, 64, 64, 64, 64, 3, NBOX, 1);
  conv(SB, nullptr, fr_w2, fr_b2, SD, 64, 64, 64, 64, 3, NBOX, 1);
  {
    int total = NBOX * 64 * 64;
    final_conv_kernel<<<dim3((total + 255) / 256), dim3(256), 0, stream>>>(SD, fr_w3, fr_b3, out);
  }
}
